// MultiHeadAttention_34548716929203
// MI455X (gfx1250) — matrix operand order repaired and checked by reading, not compiled, not run on hardware
//
#include <hip/hip_runtime.h>
#include <hip/hip_bf16.h>

// ---------------------------------------------------------------------------
// MHA forward for MI455X (gfx1250, wave32, WMMA).
//   B=2, S=2048, D=1024, H=16, HD=64
// Pipeline:
//   gemm<0>(q,Wq)->Qh[B,H,S,HD] bf16   (ws)
//   gemm<0>(k,Wk)->Kh                  (ws)
//   gemm<0>(v,Wv)->Vh                  (ws)
//   flash-attn(Qh,Kh,Vh,pad)->Ab[B,S,D] bf16 (ws)
//   gemm<1>(Ab,Wo)->d_out f32
// bf16-to-bf16 tile copies use GLOBAL_LOAD_ASYNC_TO_LDS_B128 (ASYNCcnt path)
// when the toolchain exposes the builtin.
// ---------------------------------------------------------------------------

typedef __attribute__((ext_vector_type(16))) __bf16 v16bf;
typedef __attribute__((ext_vector_type(8)))  float  v8f;
typedef int v4i __attribute__((vector_size(16)));

#define AS1 __attribute__((address_space(1)))
#define AS3 __attribute__((address_space(3)))

#if defined(__gfx1250__) && __has_builtin(__builtin_amdgcn_global_load_async_to_lds_b128)
#define USE_ASYNC_LDS 1
#else
#define USE_ASYNC_LDS 0
#endif

#if USE_ASYNC_LDS
__device__ inline void async_copy_b128(const unsigned short* g, unsigned short* l) {
    __builtin_amdgcn_global_load_async_to_lds_b128(
        (AS1 v4i*)g, (AS3 v4i*)l, /*offset=*/0, /*cpol=*/0);
}
__device__ inline void async_wait0() {
#if __has_builtin(__builtin_amdgcn_s_wait_asynccnt)
    __builtin_amdgcn_s_wait_asynccnt(0);
#else
    asm volatile("s_wait_asynccnt 0x0" ::: "memory");
#endif
}
#endif

union Frag {
    v16bf v;
    uint4 u[2];
};

__device__ inline unsigned short f2bf(float f) {
    // round-to-nearest-even fp32 -> bf16
    unsigned int u = __float_as_uint(f);
    unsigned int r = u + 0x7FFFu + ((u >> 16) & 1u);
    return (unsigned short)(r >> 16);
}

__device__ inline uint4 pack8(float4 a, float4 b) {
    union { unsigned short s[8]; uint4 u; } t;
    t.s[0] = f2bf(a.x); t.s[1] = f2bf(a.y); t.s[2] = f2bf(a.z); t.s[3] = f2bf(a.w);
    t.s[4] = f2bf(b.x); t.s[5] = f2bf(b.y); t.s[6] = f2bf(b.z); t.s[7] = f2bf(b.w);
    return t.u;
}

__device__ inline v8f wmma_bf16(const Frag& a, const Frag& b, v8f c) {
    return __builtin_amdgcn_wmma_f32_16x16x32_bf16(
        /*neg_a=*/false, a.v, /*neg_b=*/false, b.v,
        /*c_mod=*/(short)0, c, /*reuse_a=*/false, /*reuse_b=*/false);
}

__device__ inline float rmax16(float v) {
    for (int off = 8; off; off >>= 1) v = fmaxf(v, __shfl_xor(v, off, 16));
    return v;
}
__device__ inline float rsum16(float v) {
    for (int off = 8; off; off >>= 1) v += __shfl_xor(v, off, 16);
    return v;
}

// ---------------------------------------------------------------------------
// GEMM: Y[m,n] = sum_k X[m,k] * W[n,k]     M=4096, N=1024, K=1024
// MODE 0: bf16 output scattered to head-split [B,H,S,HD]
// MODE 1: f32 output, row-major [M,N]
// A_BF16: X already bf16 (attention output; async-copied to LDS); else fp32,
//         converted through VGPRs on load.
// Block tile 128x128, BK=32, 8 waves (2x4), wave tile 64x32.
// ---------------------------------------------------------------------------
template <int MODE, bool A_BF16>
__global__ __launch_bounds__(256)
void gemm_xwt(const float* __restrict__ Xf,
              const unsigned short* __restrict__ Xb,
              const float* __restrict__ W,
              unsigned short* __restrict__ Yb,
              float* __restrict__ Yf) {
    const int K = 1024;
    __shared__ unsigned short As[128 * 40];  // 128 rows, 32 k + pad
    __shared__ unsigned short Bs[128 * 40];

    const int tid  = threadIdx.x;
    const int lane = tid & 31;
    const int wave = tid >> 5;
    const int wm   = wave >> 2;   // 0..1
    const int wn   = wave & 3;    // 0..3
    const int m0   = blockIdx.x * 128;
    const int n0   = blockIdx.y * 128;

    v8f acc[4][2];
    for (int i = 0; i < 4; ++i)
        for (int j = 0; j < 2; ++j)
            acc[i][j] = (v8f){};

    const int mrow = lane & 15;
    const int kb8  = (lane >> 4) * 8;
    const int kb16 = (lane >> 4) * 16;

    for (int k0 = 0; k0 < K; k0 += 32) {
        __syncthreads();
        // cooperative tile load: 512 8-element chunks across 256 threads
        for (int u = tid; u < 512; u += 256) {
            const int r  = u >> 2;
            const int ch = (u & 3) * 8;
            // A tile
            if (A_BF16) {
#if USE_ASYNC_LDS
                async_copy_b128(Xb + (size_t)(m0 + r) * K + k0 + ch,
                                &As[r * 40 + ch]);
#else
                *(uint4*)&As[r * 40 + ch] =
                    *(const uint4*)(Xb + (size_t)(m0 + r) * K + k0 + ch);
#endif
            } else {
                const float4* p = (const float4*)(Xf + (size_t)(m0 + r) * K + k0 + ch);
                *(uint4*)&As[r * 40 + ch] = pack8(p[0], p[1]);
            }
            // B tile (rows of W along n): fp32 -> bf16 through VGPRs
            const float4* pw = (const float4*)(W + (size_t)(n0 + r) * K + k0 + ch);
            *(uint4*)&Bs[r * 40 + ch] = pack8(pw[0], pw[1]);
            // prefetch next k-step
            if (k0 + 32 < K) {
                if (!A_BF16)
                    __builtin_prefetch(Xf + (size_t)(m0 + r) * K + k0 + 32 + ch, 0, 1);
                __builtin_prefetch(W + (size_t)(n0 + r) * K + k0 + 32 + ch, 0, 1);
            }
        }
#if USE_ASYNC_LDS
        if (A_BF16) async_wait0();
#endif
        __syncthreads();

        Frag af[4], bf[2];
        for (int i = 0; i < 4; ++i) {
            const int r = wm * 64 + i * 16 + mrow;
            af[i].u[0] = *(const uint4*)&As[r * 40 + kb8];
            af[i].u[1] = *(const uint4*)&As[r * 40 + 16 + kb8];
        }
        for (int j = 0; j < 2; ++j) {
            const int r = wn * 32 + j * 16 + mrow;
            bf[j].u[0] = *(const uint4*)&Bs[r * 40 + kb8];
            bf[j].u[1] = *(const uint4*)&Bs[r * 40 + 16 + kb8];
        }
        for (int i = 0; i < 4; ++i)
            for (int j = 0; j < 2; ++j)
                acc[i][j] = wmma_bf16(af[i], bf[j], acc[i][j]);
    }

    // Epilogue. C layout: VGPR g -> local row g (lanes 0-15) or g+8 (16-31),
    // col = lane%16.
    const int rhalf = (lane >> 4) * 8;
    const int ncol  = lane & 15;
    for (int i = 0; i < 4; ++i) {
        for (int j = 0; j < 2; ++j) {
            const int mb = m0 + wm * 64 + i * 16 + rhalf;
            const int n  = n0 + wn * 32 + j * 16 + ncol;
            for (int g = 0; g < 8; ++g) {
                const int m = mb + g;
                const float val = acc[i][j][g];
                if (MODE == 0) {
                    const int b = m >> 11, s = m & 2047;
                    const int h = n >> 6,  hd = n & 63;
                    Yb[(((size_t)(b * 16 + h) * 2048 + s) << 6) + hd] = f2bf(val);
                } else {
                    Yf[(size_t)m * 1024 + n] = val;
                }
            }
        }
    }
}

// ---------------------------------------------------------------------------
// Flash attention.  Block = (q-tile of 128, head, batch); 8 waves x 16 rows.
// Streams 64-key tiles of K / V^T through LDS; K tile copied with async
// global->LDS (ASYNCcnt); online softmax; causal mask computed analytically;
// pad mask read from bool array.
// ---------------------------------------------------------------------------
__global__ __launch_bounds__(256)
void attn_kernel(const unsigned short* __restrict__ Qh,
                 const unsigned short* __restrict__ Kh,
                 const unsigned short* __restrict__ Vh,
                 const unsigned char* __restrict__ padm,
                 unsigned short* __restrict__ Ob) {
    const int S = 2048, HD = 64, H = 16;
    __shared__ unsigned short Kt[64 * 72];      // [key][hd]
    __shared__ unsigned short Vt[64 * 72];      // [hd][key]  (transposed)
    __shared__ unsigned short Pb[8 * 16 * 72];  // per-wave P rows

    const int tid  = threadIdx.x;
    const int lane = tid & 31;
    const int wave = tid >> 5;
    const int q0   = blockIdx.x * 128;
    const int h    = blockIdx.y;
    const int b    = blockIdx.z;

    const size_t hoff = (size_t)(b * H + h) * S * HD;
    const unsigned short* Qp = Qh + hoff;
    const unsigned short* Kp = Kh + hoff;
    const unsigned short* Vp = Vh + hoff;

    const int mrow  = lane & 15;
    const int kb8   = (lane >> 4) * 8;
    const int kb16  = (lane >> 4) * 16;
    const int rhalf = (lane >> 4) * 8;

    // Q fragments: 16 rows x 64 hd -> two 16x32 bf16 A-fragments
    const size_t qrow = (size_t)(q0 + wave * 16 + mrow) * HD;
    Frag qf[2];
    qf[0].u[0] = *(const uint4*)(Qp + qrow + kb8);
    qf[0].u[1] = *(const uint4*)(Qp + qrow + 16 + kb8);
    qf[1].u[0] = *(const uint4*)(Qp + qrow + 32 + kb8);
    qf[1].u[1] = *(const uint4*)(Qp + qrow + 48 + kb8);

    v8f o[4];
    for (int j = 0; j < 4; ++j) o[j] = (v8f){};
    float mi[8], li[8];
    for (int g = 0; g < 8; ++g) { mi[g] = -3.0e38f; li[g] = 0.0f; }

    const int qbase = q0 + wave * 16 + rhalf;
    const int kend  = q0 + 128;  // causal bound for this block

    for (int k0 = 0; k0 < kend; k0 += 64) {
        __syncthreads();
        // cooperative K/V tile load; K via async global->LDS, V transposed
        for (int u = tid; u < 512; u += 256) {
            const int r  = u >> 3;
            const int ch = (u & 7) * 8;
#if USE_ASYNC_LDS
            async_copy_b128(Kp + (size_t)(k0 + r) * HD + ch, &Kt[r * 72 + ch]);
#else
            *(uint4*)&Kt[r * 72 + ch] =
                *(const uint4*)(Kp + (size_t)(k0 + r) * HD + ch);
#endif
            union { uint4 u4; unsigned short s[8]; } vv;
            vv.u4 = *(const uint4*)(Vp + (size_t)(k0 + r) * HD + ch);
            for (int e = 0; e < 8; ++e) Vt[(ch + e) * 72 + r] = vv.s[e];
        }
#if USE_ASYNC_LDS
        async_wait0();
#endif
        __syncthreads();

        // S = Q K^T : 4 key n-tiles of 16, 2 chained WMMAs each (hd 0-31, 32-63)
        v8f sf[4];
        for (int j = 0; j < 4; ++j) {
            const int kr = j * 16 + mrow;  // this lane's key column
            Frag kf0, kf1;
            kf0.u[0] = *(const uint4*)&Kt[kr * 72 + kb8];
            kf0.u[1] = *(const uint4*)&Kt[kr * 72 + 16 + kb8];
            kf1.u[0] = *(const uint4*)&Kt[kr * 72 + 32 + kb8];
            kf1.u[1] = *(const uint4*)&Kt[kr * 72 + 48 + kb8];
            v8f z = (v8f){};
            z = wmma_bf16(qf[0], kf0, z);
            z = wmma_bf16(qf[1], kf1, z);
            sf[j] = z;
        }

        // scale + causal + pad masks (branch-free: no EXEC divergence)
        for (int j = 0; j < 4; ++j) {
            const int kk = k0 + j * 16 + (lane & 15);
            const bool isPad = padm[b * S + kk] != 0;
            for (int g = 0; g < 8; ++g) {
                const int qq = qbase + g;
                float sc = sf[j][g] * 0.03125f;  // 1/sqrt(1024)
                sc = (kk > qq || isPad) ? -1.0e20f : sc;
                sf[j][g] = sc;
            }
        }

        // online softmax update
        for (int g = 0; g < 8; ++g) {
            float v = fmaxf(fmaxf(sf[0][g], sf[1][g]), fmaxf(sf[2][g], sf[3][g]));
            const float mnew = fmaxf(mi[g], rmax16(v));
            const float alpha = __expf(mi[g] - mnew);
            mi[g] = mnew;
            li[g] *= alpha;
            for (int j = 0; j < 4; ++j) o[j][g] *= alpha;
            float rs = 0.0f;
            for (int j = 0; j < 4; ++j) {
                const float p = __expf(sf[j][g] - mnew);
                sf[j][g] = p;
                rs += p;
            }
            li[g] += rsum16(rs);
        }

        // P (C-frag layout) -> wave-private LDS -> A-frag layout
        unsigned short* Pw = &Pb[wave * 16 * 72];
        for (int j = 0; j < 4; ++j)
            for (int g = 0; g < 8; ++g)
                Pw[(rhalf + g) * 72 + j * 16 + (lane & 15)] = f2bf(sf[j][g]);

        Frag pf[2];
        const unsigned short* Pr = Pw + mrow * 72;
        pf[0].u[0] = *(const uint4*)(Pr + kb8);
        pf[0].u[1] = *(const uint4*)(Pr + 16 + kb8);
        pf[1].u[0] = *(const uint4*)(Pr + 32 + kb8);
        pf[1].u[1] = *(const uint4*)(Pr + 48 + kb8);

        // O += P V : 4 hd n-tiles, 2 chained WMMAs each (keys 0-31, 32-63)
        for (int j = 0; j < 4; ++j) {
            const int hd = j * 16 + mrow;  // this lane's hd column
            Frag vf0, vf1;
            vf0.u[0] = *(const uint4*)&Vt[hd * 72 + kb8];
            vf0.u[1] = *(const uint4*)&Vt[hd * 72 + 16 + kb8];
            vf1.u[0] = *(const uint4*)&Vt[hd * 72 + 32 + kb8];
            vf1.u[1] = *(const uint4*)&Vt[hd * 72 + 48 + kb8];
            o[j] = wmma_bf16(pf[0], vf0, o[j]);
            o[j] = wmma_bf16(pf[1], vf1, o[j]);
        }
    }

    // normalize and write merged-head bf16 output [B,S,D]
    for (int g = 0; g < 8; ++g) {
        const float inv = 1.0f / li[g];
        const int s = q0 + wave * 16 + rhalf + g;
        const size_t base = ((size_t)(b * 2048 + s)) * 1024 + h * 64;
        for (int j = 0; j < 4; ++j)
            Ob[base + j * 16 + (lane & 15)] = f2bf(o[j][g] * inv);
    }
}

// ---------------------------------------------------------------------------
extern "C" void kernel_launch(void* const* d_in, const int* in_sizes, int n_in,
                              void* d_out, int out_size, void* d_ws, size_t ws_size,
                              hipStream_t stream) {
    (void)in_sizes; (void)n_in; (void)out_size; (void)ws_size;
    const float* q = (const float*)d_in[0];
    const float* k = (const float*)d_in[1];
    const float* v = (const float*)d_in[2];
    const unsigned char* padm = (const unsigned char*)d_in[3];
    // d_in[4] = causal mask: computed analytically, not read
    const float* Wq = (const float*)d_in[5];
    const float* Wk = (const float*)d_in[6];
    const float* Wv = (const float*)d_in[7];
    const float* Wo = (const float*)d_in[8];

    // workspace: 4 x (4096*1024) bf16 buffers = 32 MB
    unsigned short* ws = (unsigned short*)d_ws;
    const size_t N = (size_t)4096 * 1024;
    unsigned short* Qh = ws;
    unsigned short* Kh = ws + N;
    unsigned short* Vh = ws + 2 * N;
    unsigned short* Ab = ws + 3 * N;

    dim3 gg(32, 8), bb(256);
    hipLaunchKernelGGL((gemm_xwt<0, false>), gg, bb, 0, stream,
                       q, (const unsigned short*)nullptr, Wq, Qh, (float*)nullptr);
    hipLaunchKernelGGL((gemm_xwt<0, false>), gg, bb, 0, stream,
                       k, (const unsigned short*)nullptr, Wk, Kh, (float*)nullptr);
    hipLaunchKernelGGL((gemm_xwt<0, false>), gg, bb, 0, stream,
                       v, (const unsigned short*)nullptr, Wv, Vh, (float*)nullptr);

    dim3 ga(16, 16, 2);  // (q-tiles, heads, batch)
    hipLaunchKernelGGL(attn_kernel, ga, bb, 0, stream, Qh, Kh, Vh, padm, Ab);

    hipLaunchKernelGGL((gemm_xwt<1, true>), gg, bb, 0, stream,
                       (const float*)nullptr, Ab, Wo, (unsigned short*)nullptr,
                       (float*)d_out);
}
